// SCABlock_32959579029609
// MI455X (gfx1250) — matrix operand order repaired and checked by reading, not compiled, not run on hardware
//
#include <hip/hip_runtime.h>
#include <hip/hip_bf16.h>

// ---------------------------------------------------------------------------
// Types for CDNA5 WMMA
// ---------------------------------------------------------------------------
typedef __attribute__((ext_vector_type(16))) _Float16 v16h;
typedef __attribute__((ext_vector_type(8)))  _Float16 v8h;
typedef __attribute__((ext_vector_type(8)))  float    v8f;
typedef int v4i __attribute__((ext_vector_type(4)));

#define BM 128
#define BN 128
#define BK 32
#define LDAS 40    // As row stride in halves (BK + 8) -> 80 B
#define LDBS 136   // Bs row stride in halves (BN + 8) -> 272 B

// ---------------------------------------------------------------------------
// Async global->LDS copy (GLOBAL_LOAD_ASYNC_TO_LDS_B128, ASYNCcnt) with
// graceful fallback to register copies when the builtin is unavailable.
// Builtin signature (from probe): (v4i __device__ *src, v4i __shared__ *dst,
//                                  int offset, int cpol)
// ---------------------------------------------------------------------------
#if defined(__HIP_DEVICE_COMPILE__) && defined(__has_builtin)
#if __has_builtin(__builtin_amdgcn_global_load_async_to_lds_b128)
#define HAS_ASYNC_LDS 1
#endif
#endif
#ifndef HAS_ASYNC_LDS
#define HAS_ASYNC_LDS 0
#endif

#if HAS_ASYNC_LDS
typedef __attribute__((address_space(1))) v4i gv4i;
typedef __attribute__((address_space(3))) v4i lv4i;
#endif

__device__ __forceinline__ void cp16(_Float16* ldst, const _Float16* gsrc) {
#if HAS_ASYNC_LDS
  __builtin_amdgcn_global_load_async_to_lds_b128((gv4i*)gsrc, (lv4i*)ldst, 0, 0);
#else
  *reinterpret_cast<v8h*>(ldst) = *reinterpret_cast<const v8h*>(gsrc);
#endif
}

__device__ __forceinline__ void async_copy_wait() {
#if HAS_ASYNC_LDS
  asm volatile("s_wait_asynccnt 0x0" ::: "memory");
#endif
}

// ---------------------------------------------------------------------------
// Generic f16 GEMM: C = act(alpha*(A*B + bias)) + R
//   A: f16 row-major [M x K], stride lda (halves)
//   B: f16 K-major  [K x N], stride ldb (halves)  (i.e. B[k][n] contiguous in n)
//   per z (z = b*H + h): base += b*{a,b,r,c}zb + h*{a,b,r,c}zh
//   R: fp32 (rf16=0) or f16 (rf16=1); C: f16 (cf32=0) or fp32 (cf32=1)
// Requirements: M%128==0, K%32==0, all strides/base offsets multiples of
// 8 halves (16 B) for async copies.
// ---------------------------------------------------------------------------
struct GemmP {
  const _Float16* A; long long azb, azh; int lda;
  const _Float16* B; long long bzb, bzh; int ldb;
  const float* bias;
  const void* R; long long rzb, rzh; int ldr; int rf16;
  void* C; long long czb, czh; int ldc; int cf32;
  int M, N, K, H, Z;
  float alpha; int act;   // 0 = none, 1 = exact GELU
};

__device__ __forceinline__ v16h cat8(v8h lo, v8h hi) {
  return __builtin_shufflevector(lo, hi, 0, 1, 2, 3, 4, 5, 6, 7,
                                         8, 9, 10, 11, 12, 13, 14, 15);
}

__global__ __launch_bounds__(256) void gemm_f16_wmma(GemmP p) {
  __shared__ __align__(16) _Float16 As[2][BM * LDAS];
  __shared__ __align__(16) _Float16 Bs[2][BK * LDBS];

  const int t    = threadIdx.x;
  const int wid  = t >> 5;
  const int lane = t & 31;
  const int wm   = (wid & 1) * 64;   // wave row offset in block tile
  const int wn   = (wid >> 1) * 32;  // wave col offset in block tile

  const int zb = (int)blockIdx.z / p.H;
  const int zh = (int)blockIdx.z % p.H;
  const _Float16* A = p.A + (long long)zb * p.azb + (long long)zh * p.azh;
  const _Float16* B = p.B + (long long)zb * p.bzb + (long long)zh * p.bzh;
  float* C32 = (float*)p.C + (long long)zb * p.czb + (long long)zh * p.czh;
  _Float16* C16 = (_Float16*)p.C + (long long)zb * p.czb + (long long)zh * p.czh;
  const float* R32 = (const float*)p.R;
  const _Float16* R16 = (const _Float16*)p.R;
  if (p.R) {
    R32 += (long long)zb * p.rzb + (long long)zh * p.rzh;
    R16 += (long long)zb * p.rzb + (long long)zh * p.rzh;
  }

  const int m0 = blockIdx.y * BM;
  const int n0 = blockIdx.x * BN;

  // ---- staging: pure 16B copies (async path when available) ----
  // A tile: 128 rows x 32 halves = 512 chunks of 16 B -> 2 per thread
  // B tile: 32 rows x 128 halves = 512 chunks of 16 B -> 2 per thread
  auto stageA = [&](int k0, int buf) {
#pragma unroll
    for (int i = 0; i < 2; ++i) {
      int id = t + i * 256;
      int row = id >> 2, ch = (id & 3) * 8;
      cp16(&As[buf][row * LDAS + ch],
           &A[(long long)(m0 + row) * p.lda + k0 + ch]);
    }
  };
  auto stageB = [&](int k0, int buf) {
#pragma unroll
    for (int i = 0; i < 2; ++i) {
      int id = t + i * 256;
      int kk = id >> 4, ch = (id & 15) * 8;
      cp16(&Bs[buf][kk * LDBS + ch],
           &B[(long long)(k0 + kk) * p.ldb + n0 + ch]);
    }
  };

  v8f acc[4][2];
#pragma unroll
  for (int mi = 0; mi < 4; ++mi)
#pragma unroll
    for (int ni = 0; ni < 2; ++ni)
      acc[mi][ni] = (v8f)(0.0f);

  const int g  = lane >> 4;   // lane group (0/1)
  const int ml = lane & 15;

  const int S = p.K / BK;
  stageA(0, 0);
  stageB(0, 0);
  async_copy_wait();
  __syncthreads();

  for (int s = 0; s < S; ++s) {
    const int cur  = s & 1;
    const bool more = (s + 1 < S);
    if (more) {                       // issue next tile's copies (async)
      stageA((s + 1) * BK, cur ^ 1);
      stageB((s + 1) * BK, cur ^ 1);
    }

    // ---- WMMA from buffer `cur` ----
    // A 16x32 f16: lane group g, row M=ml; e=0..7 -> K=8g+e, e=8..15 -> K=16+8g+(e-8)
    v16h af[4];
#pragma unroll
    for (int mi = 0; mi < 4; ++mi) {
      const _Float16* sp = &As[cur][(wm + mi * 16 + ml) * LDAS + g * 8];
      v8h lo = *reinterpret_cast<const v8h*>(sp);
      v8h hi = *reinterpret_cast<const v8h*>(sp + 16);
      af[mi] = cat8(lo, hi);
    }
    // B 32x16 f16: lane group g, col N=ml; e=0..7 -> K=8g+e, e=8..15 -> K=16+8g+(e-8)
    v16h bf[2];
#pragma unroll
    for (int ni = 0; ni < 2; ++ni) {
      const _Float16* sp = &Bs[cur][wn + ni * 16 + ml];
#pragma unroll
      for (int e = 0; e < 16; ++e) {
        int kk = (e < 8) ? (g * 8 + e) : (16 + g * 8 + (e - 8));
        bf[ni][e] = sp[kk * LDBS];
      }
    }
#pragma unroll
    for (int mi = 0; mi < 4; ++mi)
#pragma unroll
      for (int ni = 0; ni < 2; ++ni)
        acc[mi][ni] = __builtin_amdgcn_wmma_f32_16x16x32_f16(
            false, af[mi], false, bf[ni], (short)0, acc[mi][ni], false, false);

    if (more) async_copy_wait();      // own copies landed before barrier
    __syncthreads();
  }

  // ---- epilogue: C f32 layout: lane -> (N = ml, M = r + 8*g) ----
#pragma unroll
  for (int ni = 0; ni < 2; ++ni) {
    int colg = n0 + wn + ni * 16 + ml;
    if (colg < p.N) {
      float bv = p.bias ? p.bias[colg] : 0.0f;
#pragma unroll
      for (int mi = 0; mi < 4; ++mi) {
#pragma unroll
        for (int r = 0; r < 8; ++r) {
          int rowg = m0 + wm + mi * 16 + g * 8 + r;
          float v = p.alpha * (acc[mi][ni][r] + bv);
          if (p.act == 1) v = 0.5f * v * (1.0f + erff(v * 0.70710678118654752f));
          if (p.R) {
            long long ri = (long long)rowg * p.ldr + colg;
            v += p.rf16 ? (float)R16[ri] : R32[ri];
          }
          long long ci = (long long)rowg * p.ldc + colg;
          if (p.cf32) C32[ci] = v;
          else        C16[ci] = (_Float16)v;
        }
      }
    }
  }
}

// ---------------------------------------------------------------------------
// LayerNorm: one 256-thread block per row of length D, f16 output
// ---------------------------------------------------------------------------
template <typename T>
__global__ __launch_bounds__(256) void layernorm_k(const T* __restrict__ x,
                                                   const float* __restrict__ gam,
                                                   const float* __restrict__ bet,
                                                   _Float16* __restrict__ y, int D) {
  __shared__ float sh[256];
  const T* xr = x + (long long)blockIdx.x * D;
  _Float16* yr = y + (long long)blockIdx.x * D;

  float s = 0.0f;
  for (int i = threadIdx.x; i < D; i += 256) s += (float)xr[i];
  sh[threadIdx.x] = s; __syncthreads();
  for (int st = 128; st > 0; st >>= 1) {
    if (threadIdx.x < st) sh[threadIdx.x] += sh[threadIdx.x + st];
    __syncthreads();
  }
  float mean = sh[0] / (float)D; __syncthreads();

  float s2 = 0.0f;
  for (int i = threadIdx.x; i < D; i += 256) {
    float d = (float)xr[i] - mean; s2 += d * d;
  }
  sh[threadIdx.x] = s2; __syncthreads();
  for (int st = 128; st > 0; st >>= 1) {
    if (threadIdx.x < st) sh[threadIdx.x] += sh[threadIdx.x + st];
    __syncthreads();
  }
  float rstd = rsqrtf(sh[0] / (float)D + 1e-5f); __syncthreads();

  for (int i = threadIdx.x; i < D; i += 256)
    yr[i] = (_Float16)(((float)xr[i] - mean) * rstd * gam[i] + bet[i]);
}

// ---------------------------------------------------------------------------
// Row softmax (in place, f16 storage / f32 math)
// ---------------------------------------------------------------------------
__global__ __launch_bounds__(256) void softmax_k(_Float16* __restrict__ x, int n) {
  __shared__ float sh[256];
  _Float16* xr = x + (long long)blockIdx.x * n;

  float mx = -3.4e38f;
  for (int i = threadIdx.x; i < n; i += 256) mx = fmaxf(mx, (float)xr[i]);
  sh[threadIdx.x] = mx; __syncthreads();
  for (int st = 128; st > 0; st >>= 1) {
    if (threadIdx.x < st) sh[threadIdx.x] = fmaxf(sh[threadIdx.x], sh[threadIdx.x + st]);
    __syncthreads();
  }
  mx = sh[0]; __syncthreads();

  float s = 0.0f;
  for (int i = threadIdx.x; i < n; i += 256) {
    float e = __expf((float)xr[i] - mx); xr[i] = (_Float16)e; s += e;
  }
  sh[threadIdx.x] = s; __syncthreads();
  for (int st = 128; st > 0; st >>= 1) {
    if (threadIdx.x < st) sh[threadIdx.x] += sh[threadIdx.x + st];
    __syncthreads();
  }
  float inv = 1.0f / sh[0];
  for (int i = threadIdx.x; i < n; i += 256)
    xr[i] = (_Float16)((float)xr[i] * inv);
}

// ---------------------------------------------------------------------------
// small utility kernels
// ---------------------------------------------------------------------------
// fp32 -> f16 elementwise
__global__ void conv_h(const float* __restrict__ s, _Float16* __restrict__ d,
                       long long n) {
  long long i = (long long)blockIdx.x * blockDim.x + threadIdx.x;
  if (i < n) d[i] = (_Float16)s[i];
}
// weight [N,K] fp32 -> [K,N] f16 (K-major for GEMM B input)
__global__ void tconv_w(const float* __restrict__ Wf, _Float16* __restrict__ wt,
                        int N, int K) {
  long long i = (long long)blockIdx.x * blockDim.x + threadIdx.x;
  if (i >= (long long)N * K) return;
  int n = (int)(i / K), k = (int)(i % K);
  wt[(long long)k * N + n] = (_Float16)Wf[i];
}
// strided f16 transpose: dst[c*ldd + r] = src[r*lds + c]
__global__ void transpose_h(const _Float16* __restrict__ src, int ldsrc,
                            _Float16* __restrict__ dst, long long lddst,
                            long long rows, int cols) {
  long long i = (long long)blockIdx.x * blockDim.x + threadIdx.x;
  if (i >= rows * cols) return;
  long long r = i / cols; int c = (int)(i % cols);
  dst[(long long)c * lddst + r] = src[r * (long long)ldsrc + c];
}
// f16 concat along last dim
__global__ void concat_k(const _Float16* __restrict__ a,
                         const _Float16* __restrict__ b,
                         _Float16* __restrict__ o, long long rows, int d) {
  long long i = (long long)blockIdx.x * blockDim.x + threadIdx.x;
  long long total = rows * 2LL * d;
  if (i >= total) return;
  long long r = i / (2 * d);
  int c = (int)(i % (2 * d));
  o[i] = (c < d) ? a[r * d + c] : b[r * d + (c - d)];
}

static inline void run_gemm(hipStream_t s, GemmP p) {
  dim3 grid((p.N + BN - 1) / BN, (p.M + BM - 1) / BM, p.Z);
  gemm_f16_wmma<<<grid, dim3(256), 0, s>>>(p);
}
static inline unsigned cdiv(long long a, long long b) {
  return (unsigned)((a + b - 1) / b);
}

// ---------------------------------------------------------------------------
extern "C" void kernel_launch(void* const* d_in, const int* in_sizes, int n_in,
                              void* d_out, int out_size, void* d_ws, size_t ws_size,
                              hipStream_t stream) {
  (void)in_sizes; (void)n_in; (void)out_size; (void)ws_size;

  const float* motion      = (const float*)d_in[0];
  const float* scene_feats = (const float*)d_in[2];
  const float* prior       = (const float*)d_in[3];
  const float* ln1g = (const float*)d_in[4];
  const float* ln1b = (const float*)d_in[5];
  const float* qkvW = (const float*)d_in[6];
  const float* qkvB = (const float*)d_in[7];
  const float* outW = (const float*)d_in[8];
  const float* outB = (const float*)d_in[9];
  const float* ln2g = (const float*)d_in[10];
  const float* ln2b = (const float*)d_in[11];
  const float* ff1W = (const float*)d_in[12];
  const float* ff1B = (const float*)d_in[13];
  const float* ff2W = (const float*)d_in[14];
  const float* ff2B = (const float*)d_in[15];
  const float* saqW = (const float*)d_in[16];
  const float* saqB = (const float*)d_in[17];
  const float* kvW  = (const float*)d_in[18];
  const float* kvB  = (const float*)d_in[19];
  const float* saoW = (const float*)d_in[20];
  const float* saoB = (const float*)d_in[21];
  const float* fc1W = (const float*)d_in[22];
  const float* fc1B = (const float*)d_in[23];
  const float* fc2W = (const float*)d_in[24];
  const float* fc2B = (const float*)d_in[25];
  float* outp = (float*)d_out;

  // ---- workspace carve (halves) ----
  _Float16* ws = (_Float16*)d_ws;
  size_t off = 0;
  auto alloc = [&](size_t n) { _Float16* p = ws + off; off += n; return p; };
  _Float16* hA   = alloc(4096 * 512);            // ln1 / h2 / saq / sm
  _Float16* hB   = alloc(4096 * 512);            // enc_o / ff1 / smf
  _Float16* hC   = alloc(4096 * 512);            // x
  _Float16* hD   = alloc(4096 * 512);            // motion_sca
  _Float16* qkv  = alloc(4096 * 1536);           // encoder qkv (row-major)
  _Float16* kv   = alloc(32768LL * 1024);        // SA kv (row-major)
  _Float16* att  = alloc(8LL * 4 * 512 * 4096);  // enc att then SA att
  _Float16* kT   = alloc(512LL * 32768);         // transposed K (enc then SA)
  _Float16* catb = alloc(4096 * 1024);
  _Float16* fc1b = alloc(4096 * 2048);
  _Float16* sc16 = alloc(32768LL * 512);         // scene_feats f16
  _Float16* wqkv = alloc(512 * 1536);
  _Float16* wout = alloc(512 * 512);
  _Float16* wff1 = alloc(512 * 512);
  _Float16* wff2 = alloc(512 * 512);
  _Float16* wsaq = alloc(512 * 512);
  _Float16* wkv  = alloc(512 * 1024);
  _Float16* wsao = alloc(512 * 512);
  _Float16* wfc1 = alloc(1024 * 2048);
  _Float16* wfc2 = alloc(2048 * 512);

  // ---- one-time conversions (per call) ----
  conv_h<<<cdiv(32768LL * 512, 256), 256, 0, stream>>>(scene_feats, sc16,
                                                       32768LL * 512);
  tconv_w<<<cdiv(1536 * 512, 256), 256, 0, stream>>>(qkvW, wqkv, 1536, 512);
  tconv_w<<<cdiv(512 * 512, 256),  256, 0, stream>>>(outW, wout, 512, 512);
  tconv_w<<<cdiv(512 * 512, 256),  256, 0, stream>>>(ff1W, wff1, 512, 512);
  tconv_w<<<cdiv(512 * 512, 256),  256, 0, stream>>>(ff2W, wff2, 512, 512);
  tconv_w<<<cdiv(512 * 512, 256),  256, 0, stream>>>(saqW, wsaq, 512, 512);
  tconv_w<<<cdiv(1024 * 512, 256), 256, 0, stream>>>(kvW, wkv, 1024, 512);
  tconv_w<<<cdiv(512 * 512, 256),  256, 0, stream>>>(saoW, wsao, 512, 512);
  tconv_w<<<cdiv(2048 * 1024, 256), 256, 0, stream>>>(fc1W, wfc1, 2048, 1024);
  tconv_w<<<cdiv(512 * 2048, 256), 256, 0, stream>>>(fc2W, wfc2, 512, 2048);

  GemmP g0{}; g0.H = 1; g0.Z = 1; g0.alpha = 1.0f;

  // 1) ln1 = LN(motion)  (fp32 in, f16 out)
  layernorm_k<float><<<4096, 256, 0, stream>>>(motion, ln1g, ln1b, hA, 512);

  // 2) qkv = ln1 @ qkvW^T + b   [4096 x 1536]
  { GemmP p = g0;
    p.A = hA; p.lda = 512;
    p.B = wqkv; p.ldb = 1536;
    p.bias = qkvB;
    p.C = qkv; p.ldc = 1536;
    p.M = 4096; p.N = 1536; p.K = 512;
    run_gemm(stream, p); }

  // 3) kT = transpose of encoder K-slice: [512 feat x 4096 pos]
  transpose_h<<<cdiv(4096LL * 512, 256), 256, 0, stream>>>(qkv + 512, 1536,
                                                           kT, 4096, 4096, 512);

  // 4) encoder scores: att[b,h] = (Q @ K^T) * 1/sqrt(64)   (Z=64, H=8)
  { GemmP p = g0;
    p.H = 8; p.Z = 64; p.alpha = 0.125f;
    p.A = qkv; p.azb = 786432;  p.azh = 64;     p.lda = 1536;   // Q slice
    p.B = kT;  p.bzb = 512;     p.bzh = 262144; p.ldb = 4096;   // K^T
    p.C = att; p.czb = 2097152; p.czh = 262144; p.ldc = 512;
    p.M = 512; p.N = 512; p.K = 64;
    run_gemm(stream, p); }

  // 5) softmax over 512 keys
  softmax_k<<<64 * 512, 256, 0, stream>>>(att, 512);

  // 6) enc_o[b,h] = att @ V   (V slice of qkv is already K-major: B[k=pos][n=feat])
  { GemmP p = g0;
    p.H = 8; p.Z = 64;
    p.A = att;        p.azb = 2097152; p.azh = 262144; p.lda = 512;
    p.B = qkv + 1024; p.bzb = 786432;  p.bzh = 64;     p.ldb = 1536;
    p.C = hB;         p.czb = 262144;  p.czh = 64;     p.ldc = 512;
    p.M = 512; p.N = 64; p.K = 512;
    run_gemm(stream, p); }

  // 7) x = motion + enc_o @ outW^T + b
  { GemmP p = g0;
    p.A = hB; p.lda = 512;
    p.B = wout; p.ldb = 512;
    p.bias = outB;
    p.R = motion; p.ldr = 512; p.rf16 = 0;
    p.C = hC; p.ldc = 512;
    p.M = 4096; p.N = 512; p.K = 512;
    run_gemm(stream, p); }

  // 8) h2 = LN(x)  (f16 in, f16 out)
  layernorm_k<_Float16><<<4096, 256, 0, stream>>>(hC, ln2g, ln2b, hA, 512);

  // 9) ff1 = gelu(h2 @ ff1W^T + b)
  { GemmP p = g0;
    p.A = hA; p.lda = 512;
    p.B = wff1; p.ldb = 512;
    p.bias = ff1B; p.act = 1;
    p.C = hB; p.ldc = 512;
    p.M = 4096; p.N = 512; p.K = 512;
    run_gemm(stream, p); }

  // 10) motion_sca = x + ff1 @ ff2W^T + b   (f16 residual)
  { GemmP p = g0;
    p.A = hB; p.lda = 512;
    p.B = wff2; p.ldb = 512;
    p.bias = ff2B;
    p.R = hC; p.ldr = 512; p.rf16 = 1;
    p.C = hD; p.ldc = 512;
    p.M = 4096; p.N = 512; p.K = 512;
    run_gemm(stream, p); }

  // 11) saq = (motion_sca @ saqW^T + b) * D^-0.5
  { GemmP p = g0;
    p.A = hD; p.lda = 512;
    p.B = wsaq; p.ldb = 512;
    p.bias = saqB; p.alpha = 0.044194173824159216f; // 1/sqrt(512)
    p.C = hA; p.ldc = 512;
    p.M = 4096; p.N = 512; p.K = 512;
    run_gemm(stream, p); }

  // 12) kv = scene_feats @ kvW^T + b   [32768 x 1024]
  { GemmP p = g0;
    p.A = sc16; p.lda = 512;
    p.B = wkv; p.ldb = 1024;
    p.bias = kvB;
    p.C = kv; p.ldc = 1024;
    p.M = 32768; p.N = 1024; p.K = 512;
    run_gemm(stream, p); }

  // 13) kT = transpose of SA K-slice: [512 feat x 32768 pos]
  transpose_h<<<cdiv(32768LL * 512, 256), 256, 0, stream>>>(kv, 1024,
                                                            kT, 32768, 32768, 512);

  // 14) SA scores: att[b,h] = Q @ K^T + prior   (Z=32, H=4)
  //     faithful torch reshape: Q of (b,h) = contiguous [512 x 128] chunk of saq
  { GemmP p = g0;
    p.H = 4; p.Z = 32;
    p.A = hA;    p.azb = 262144;  p.azh = 65536;   p.lda = 128;
    p.B = kT;    p.bzb = 4096;    p.bzh = 4194304; p.ldb = 32768;
    p.R = prior; p.rzb = 2097152; p.rzh = 0;       p.ldr = 4096; p.rf16 = 0;
    p.C = att;   p.czb = 8388608; p.czh = 2097152; p.ldc = 4096;
    p.M = 512; p.N = 4096; p.K = 128;
    run_gemm(stream, p); }

  // 15) softmax over 4096 keys
  softmax_k<<<32 * 512, 256, 0, stream>>>(att, 4096);

  // 16) sm[b,h] = att @ V   (V slice of kv is K-major: B[k=pos][n=feat])
  { GemmP p = g0;
    p.H = 4; p.Z = 32;
    p.A = att;      p.azb = 8388608; p.azh = 2097152; p.lda = 4096;
    p.B = kv + 512; p.bzb = 4194304; p.bzh = 128;     p.ldb = 1024;
    p.C = hA;       p.czb = 262144;  p.czh = 128;     p.ldc = 512;
    p.M = 512; p.N = 128; p.K = 4096;
    run_gemm(stream, p); }

  // 17) smf = sm @ saoW^T + b
  { GemmP p = g0;
    p.A = hA; p.lda = 512;
    p.B = wsao; p.ldb = 512;
    p.bias = saoB;
    p.C = hB; p.ldc = 512;
    p.M = 4096; p.N = 512; p.K = 512;
    run_gemm(stream, p); }

  // 18) cat = [motion_sca, smf]   [4096 x 1024]
  concat_k<<<cdiv(4096LL * 1024, 256), 256, 0, stream>>>(hD, hB, catb,
                                                         4096LL, 512);

  // 19) fc1 = gelu(cat @ fc1W^T + b)   [4096 x 2048]
  { GemmP p = g0;
    p.A = catb; p.lda = 1024;
    p.B = wfc1; p.ldb = 2048;
    p.bias = fc1B; p.act = 1;
    p.C = fc1b; p.ldc = 2048;
    p.M = 4096; p.N = 2048; p.K = 1024;
    run_gemm(stream, p); }

  // 20) out = fc1 @ fc2W^T + b   [4096 x 512] -> d_out (fp32)
  { GemmP p = g0;
    p.A = fc1b; p.lda = 2048;
    p.B = wfc2; p.ldb = 512;
    p.bias = fc2B;
    p.C = outp; p.ldc = 512; p.cf32 = 1;
    p.M = 4096; p.N = 512; p.K = 2048;
    run_gemm(stream, p); }
}
